// PositionEmbeddingLayer_27127013441596
// MI455X (gfx1250) — matrix operand order repaired and checked by reading, not compiled, not run on hardware
//
#include <hip/hip_runtime.h>
#include <math.h>

#define Bq 2
#define Mq 25000
#define Nq 100000
#define Kq 16
#define GN_EPS 1e-5f
#define GRIDB 512

typedef __attribute__((ext_vector_type(16))) _Float16 v16h;
typedef __attribute__((ext_vector_type(8)))  _Float16 h8;
typedef __attribute__((ext_vector_type(8)))  float    v8f;

__device__ __forceinline__ void wbar() { __builtin_amdgcn_wave_barrier(); }

// ---------------------------------------------------------------------------
// Pass 1: per-batch moments of local_pattern lp[4]:
//   stats1[b] = { sum lp_i (4), sum lp_i*lp_j upper-tri (10) }
// ---------------------------------------------------------------------------
__global__ void k_moments(const float* __restrict__ q, const float* __restrict__ kx,
                          const int* __restrict__ idx, float* __restrict__ stats1) {
  __shared__ float ls[28];
  int t = threadIdx.x;
  if (t < 28) ls[t] = 0.f;
  __syncthreads();
  int e = blockIdx.x * 256 + t;
  if (e < Bq * Mq * Kq) {
    int b  = e / (Mq * Kq);
    int mk = e - b * (Mq * Kq);
    int m  = mk >> 4;
    int id = idx[e];
    float qx = q[(b*3+0)*Mq + m];
    float qy = q[(b*3+1)*Mq + m];
    float qz = q[(b*3+2)*Mq + m];
    float ox = kx[(b*3+0)*Nq + id] - qx;
    float oy = kx[(b*3+1)*Nq + id] - qy;
    float oz = kx[(b*3+2)*Nq + id] - qz;
    float d   = sqrtf(ox*ox + oy*oy + oz*oz);
    float inv = 1.f / fmaxf(d, 1e-12f);
    float lp[4] = {ox*inv, oy*inv, oz*inv, d};
    float* base = ls + b*14;
    #pragma unroll
    for (int i = 0; i < 4; ++i) atomicAdd(&base[i], lp[i]);
    int p = 4;
    #pragma unroll
    for (int i = 0; i < 4; ++i)
      #pragma unroll
      for (int j = i; j < 4; ++j) { atomicAdd(&base[p], lp[i]*lp[j]); ++p; }
  }
  __syncthreads();
  if (t < 28) atomicAdd(&stats1[t], ls[t]);
}

// ---------------------------------------------------------------------------
// GN1 finalize: per-channel affine (scale, shift) from lp moments + W1, b1.
// h1_c = w_c . lp + b_c  =>  E[h1_c] = w_c.m + b_c ;  E[h1_c^2] = w_c^T S w_c + 2 b_c (w_c.m) + b_c^2
// group stats = mean over 8 channels (equal counts => exact).
// ---------------------------------------------------------------------------
__global__ void k_gn1(const float* __restrict__ stats1, const float* __restrict__ w1,
                      const float* __restrict__ b1, const float* __restrict__ g,
                      const float* __restrict__ be,
                      float* __restrict__ scale, float* __restrict__ shift) {
  __shared__ float mu[128], e2[128];
  int t = threadIdx.x;          // 0..127 = b*64 + c
  int b = t >> 6, c = t & 63;
  const float invn = 1.f / (float)(Mq * Kq);
  const float* st = stats1 + b*14;
  float m[4];
  #pragma unroll
  for (int i = 0; i < 4; ++i) m[i] = st[i] * invn;
  float S[4][4];
  int p = 4;
  #pragma unroll
  for (int i = 0; i < 4; ++i)
    #pragma unroll
    for (int j = i; j < 4; ++j) { float v = st[p]*invn; ++p; S[i][j]=v; S[j][i]=v; }
  float w[4];
  #pragma unroll
  for (int i = 0; i < 4; ++i) w[i] = w1[c*4+i];
  float wm = w[0]*m[0] + w[1]*m[1] + w[2]*m[2] + w[3]*m[3];
  float bc = b1[c];
  float quad = 0.f;
  #pragma unroll
  for (int i = 0; i < 4; ++i)
    #pragma unroll
    for (int j = 0; j < 4; ++j) quad += w[i]*w[j]*S[i][j];
  mu[t] = wm + bc;
  e2[t] = quad + 2.f*bc*wm + bc*bc;
  __syncthreads();
  int gb = t & ~7;
  float gm = 0.f, ge = 0.f;
  #pragma unroll
  for (int k = 0; k < 8; ++k) { gm += mu[gb+k]; ge += e2[gb+k]; }
  gm *= 0.125f; ge *= 0.125f;
  float var = ge - gm*gm;
  float sc  = g[c] * rsqrtf(var + GN_EPS);
  scale[t] = sc;
  shift[t] = be[c] - gm*sc;
}

// ---------------------------------------------------------------------------
// GN2 finalize: from accumulated per-channel sum/sumsq of h2.
// ---------------------------------------------------------------------------
__global__ void k_gn2(const float* __restrict__ stats2, const float* __restrict__ g,
                      const float* __restrict__ be,
                      float* __restrict__ scale, float* __restrict__ shift) {
  int t = threadIdx.x;          // 0..127
  int b = t >> 6, c = t & 63;
  int gb = c & ~7;
  float s = 0.f, q2 = 0.f;
  #pragma unroll
  for (int k = 0; k < 8; ++k) {
    s  += stats2[(b*64 + gb + k)*2 + 0];
    q2 += stats2[(b*64 + gb + k)*2 + 1];
  }
  const float invn = 1.f / (8.f * Mq * Kq);
  float mu = s * invn, var = q2*invn - mu*mu;
  float sc = g[c] * rsqrtf(var + GN_EPS);
  scale[t] = sc;
  shift[t] = be[c] - mu*sc;
}

// ---------------------------------------------------------------------------
// WMMA helpers. LDS layouts:
//   Weights: row-major [64 rows][72 halves] (16B pad per row for bank spread)
//   Activation tile X: [16 cols][72 halves] (col-major columns, channel = K)
// A-frag (16x32 f16): lane l -> row r*16+(l&15); K runs {0..7,16..23} (+8 if l>=16)
// B-frag (32x16 f16): lane l -> col l&15; K runs {0..7,16..23} (+8 if l>=16)
// C/D (16x16 f32): lane l -> col l&15, rows base+(l>=16?8:0)+v
// ---------------------------------------------------------------------------
__device__ __forceinline__ v16h load_afrag(const _Float16* W, int r, int kc, int lane) {
  int row  = r*16 + (lane & 15);
  int koff = kc*32 + ((lane & 16) ? 8 : 0);
  const _Float16* p = W + row*72 + koff;
  v16h a;
  ((h8*)&a)[0] = *(const h8*)(p);
  ((h8*)&a)[1] = *(const h8*)(p + 16);
  return a;
}
__device__ __forceinline__ v16h load_bfrag(const _Float16* X, int kc, int lane) {
  const _Float16* p = X + (lane & 15)*72 + kc*32 + ((lane & 16) ? 8 : 0);
  v16h b;
  ((h8*)&b)[0] = *(const h8*)(p);
  ((h8*)&b)[1] = *(const h8*)(p + 16);
  return b;
}
__device__ __forceinline__ v8f gemm64(const _Float16* W, const _Float16* X,
                                      const float* bias, int r, int lane) {
  int rbase = r*16 + ((lane & 16) ? 8 : 0);
  v8f c;
  #pragma unroll
  for (int v = 0; v < 8; ++v) c[v] = bias[rbase + v];
  v16h a0 = load_afrag(W, r, 0, lane);
  v16h b0 = load_bfrag(X, 0, lane);
  c = __builtin_amdgcn_wmma_f32_16x16x32_f16(false, a0, false, b0, (short)0, c, false, false);
  v16h a1 = load_afrag(W, r, 1, lane);
  v16h b1 = load_bfrag(X, 1, lane);
  c = __builtin_amdgcn_wmma_f32_16x16x32_f16(false, a1, false, b1, (short)0, c, false, false);
  return c;
}

// ---------------------------------------------------------------------------
// Fused pipeline. One wave = one query point m (16 columns = its 16 neighbors).
// FINAL=false: run through h2, accumulate GN2 channel stats -> outp = stats2
// FINAL=true : full pipeline + softmax + weighted sum      -> outp = output
// ---------------------------------------------------------------------------
template <bool FINAL>
__global__ void k_pipe(const float* __restrict__ q_xyzs, const float* __restrict__ k_xyzs,
                       const int* __restrict__ knn, const unsigned char* __restrict__ mask,
                       const float* __restrict__ pe_w1, const float* __restrict__ pe_b1,
                       const float* __restrict__ pe_w2, const float* __restrict__ pe_b2,
                       const float* __restrict__ at_w1, const float* __restrict__ at_b1,
                       const float* __restrict__ at_w2, const float* __restrict__ at_b2,
                       const float* __restrict__ g1s_, const float* __restrict__ g1t_,
                       const float* __restrict__ g2s_, const float* __restrict__ g2t_,
                       float* __restrict__ outp) {
  extern __shared__ char smem[];
  _Float16* wPE2 = (_Float16*)smem;
  _Float16* wAT1 = wPE2 + 64*72;
  _Float16* wAT2 = wAT1 + 64*72;
  float* fp  = (float*)(wAT2 + 64*72);
  float* w1s = fp; fp += 256;
  float* b1s = fp; fp += 64;
  float* b2pe = fp; fp += 64;
  float* b1at = fp; fp += 64;
  float* b2at = fp; fp += 64;
  float* s1 = fp; fp += 64;
  float* t1 = fp; fp += 64;
  float* s2 = fp; fp += 64;
  float* t2 = fp; fp += 64;
  float* stat = fp; fp += 128;
  char* wb0 = (char*)fp;

  const int bb   = (blockIdx.x >= GRIDB) ? 1 : 0;
  const int blk  = blockIdx.x - bb*GRIDB;
  const int wave = threadIdx.x >> 5;
  const int lane = threadIdx.x & 31;

  constexpr int WBYTES = FINAL ? (2*16*72*2 + 2*64*18*4) : (2*16*72*2);
  _Float16* xa = (_Float16*)(wb0 + wave*WBYTES);
  _Float16* xb = xa + 16*72;
  float* pef = (float*)(xb + 16*72);
  float* atn = pef + 64*18;

  // stage weights (f32 -> f16) and per-batch GN affines into LDS
  for (int i = threadIdx.x; i < 4096; i += 256) {
    int r = i >> 6, c = i & 63;
    wPE2[r*72+c] = (_Float16)pe_w2[i];
    wAT1[r*72+c] = (_Float16)at_w1[i];
    wAT2[r*72+c] = (_Float16)at_w2[i];
  }
  if (threadIdx.x < 256) w1s[threadIdx.x] = pe_w1[threadIdx.x];
  if (threadIdx.x < 64) {
    int c = threadIdx.x;
    b1s[c]  = pe_b1[c]; b2pe[c] = pe_b2[c]; b1at[c] = at_b1[c]; b2at[c] = at_b2[c];
    s1[c] = g1s_[bb*64+c]; t1[c] = g1t_[bb*64+c];
    s2[c] = g2s_[bb*64+c]; t2[c] = g2t_[bb*64+c];
  }
  if (threadIdx.x < 128) stat[threadIdx.x] = 0.f;
  __syncthreads();

  const int col = lane & 15;
  const int hi  = lane >> 4;
  float accS[32], accQ[32];
  if constexpr (!FINAL) {
    #pragma unroll
    for (int i = 0; i < 32; ++i) { accS[i] = 0.f; accQ[i] = 0.f; }
  }

  for (int m = blk*8 + wave; m < Mq; m += GRIDB*8) {
    // ---- stage 0: gather + h1 (64x4 conv) + GN1 affine + relu -> xa (f16) ----
    int e  = (bb*Mq + m)*Kq + col;
    int id = knn[e];
    float qx = q_xyzs[(bb*3+0)*Mq + m];
    float qy = q_xyzs[(bb*3+1)*Mq + m];
    float qz = q_xyzs[(bb*3+2)*Mq + m];
    float ox = k_xyzs[(bb*3+0)*Nq + id] - qx;
    float oy = k_xyzs[(bb*3+1)*Nq + id] - qy;
    float oz = k_xyzs[(bb*3+2)*Nq + id] - qz;
    float d   = sqrtf(ox*ox + oy*oy + oz*oz);
    float inv = 1.f / fmaxf(d, 1e-12f);
    float lp0 = ox*inv, lp1 = oy*inv, lp2 = oz*inv, lp3 = d;
    #pragma unroll
    for (int j = 0; j < 32; ++j) {
      int c = hi*32 + j;
      float h = w1s[c*4+0]*lp0 + w1s[c*4+1]*lp1 + w1s[c*4+2]*lp2 + w1s[c*4+3]*lp3 + b1s[c];
      h = fmaxf(h*s1[c] + t1[c], 0.f);
      xa[col*72 + c] = (_Float16)h;
    }
    wbar();
    // ---- GEMM1: pe = wPE2 @ x1 + b2pe  -> xb (f16) [+ pef (f32)] ----
    #pragma unroll
    for (int r = 0; r < 4; ++r) {
      v8f dd = gemm64(wPE2, xa, b2pe, r, lane);
      int rb = r*16 + ((lane & 16) ? 8 : 0);
      h8 hv;
      #pragma unroll
      for (int v = 0; v < 8; ++v) hv[v] = (_Float16)dd[v];
      *(h8*)(xb + col*72 + rb) = hv;
      if constexpr (FINAL) {
        #pragma unroll
        for (int v = 0; v < 8; ++v) pef[(rb+v)*18 + col] = dd[v];
      }
    }
    wbar();
    // ---- GEMM2: h2 = wAT1 @ pe + b1at ----
    #pragma unroll
    for (int r = 0; r < 4; ++r) {
      v8f dd = gemm64(wAT1, xb, b1at, r, lane);
      int rb = r*16 + ((lane & 16) ? 8 : 0);
      if constexpr (!FINAL) {
        #pragma unroll
        for (int v = 0; v < 8; ++v) { accS[r*8+v] += dd[v]; accQ[r*8+v] += dd[v]*dd[v]; }
      } else {
        h8 hv;
        #pragma unroll
        for (int v = 0; v < 8; ++v) {
          int c = rb + v;
          hv[v] = (_Float16)fmaxf(dd[v]*s2[c] + t2[c], 0.f);
        }
        *(h8*)(xa + col*72 + rb) = hv;   // x2 overwrites x1 buffer
      }
    }
    if constexpr (FINAL) {
      wbar();
      // ---- GEMM3: attn = wAT2 @ x2 + b2at -> atn (f32 [ch][col]) ----
      #pragma unroll
      for (int r = 0; r < 4; ++r) {
        v8f dd = gemm64(wAT2, xa, b2at, r, lane);
        int rb = r*16 + ((lane & 16) ? 8 : 0);
        #pragma unroll
        for (int v = 0; v < 8; ++v) atn[(rb+v)*18 + col] = dd[v];
      }
      wbar();
      // ---- masked softmax over K + weighted sum; lane handles 2 channels ----
      const unsigned char* mrow = mask + (bb*Mq + m)*Kq;
      #pragma unroll
      for (int cc = 0; cc < 2; ++cc) {
        int ch = lane*2 + cc;
        float vb[16];
        float mx = -3.402823466e38f;
        #pragma unroll
        for (int k2 = 0; k2 < 16; ++k2) {
          float a = atn[ch*18 + k2];
          a = mrow[k2] ? a : -3.402823466e38f;
          vb[k2] = a;
          mx = fmaxf(mx, a);
        }
        float s = 0.f;
        #pragma unroll
        for (int k2 = 0; k2 < 16; ++k2) { float ev = __expf(vb[k2]-mx); vb[k2]=ev; s+=ev; }
        float invs = 1.f / s;
        float o = 0.f;
        #pragma unroll
        for (int k2 = 0; k2 < 16; ++k2) o += pef[ch*18 + k2]*vb[k2];
        outp[(bb*64 + ch)*Mq + m] = o*invs;
      }
    }
    wbar();
  }

  if constexpr (!FINAL) {
    // flush per-lane h2 stats: LDS reduce then global atomics
    #pragma unroll
    for (int i = 0; i < 32; ++i) {
      int ch = (i>>3)*16 + ((lane & 16) ? 8 : 0) + (i & 7);
      atomicAdd(&stat[ch*2+0], accS[i]);
      atomicAdd(&stat[ch*2+1], accQ[i]);
    }
    __syncthreads();
    if (threadIdx.x < 128) atomicAdd(&outp[bb*128 + threadIdx.x], stat[threadIdx.x]);
  }
}

// ---------------------------------------------------------------------------
extern "C" void kernel_launch(void* const* d_in, const int* in_sizes, int n_in,
                              void* d_out, int out_size, void* d_ws, size_t ws_size,
                              hipStream_t stream) {
  const float* q_xyzs = (const float*)d_in[0];
  const float* k_xyzs = (const float*)d_in[1];
  const int*   knn    = (const int*)d_in[2];
  const unsigned char* mask = (const unsigned char*)d_in[3];
  const float* pe_w1 = (const float*)d_in[4];
  const float* pe_b1 = (const float*)d_in[5];
  const float* pe_g  = (const float*)d_in[6];
  const float* pe_be = (const float*)d_in[7];
  const float* pe_w2 = (const float*)d_in[8];
  const float* pe_b2 = (const float*)d_in[9];
  const float* at_w1 = (const float*)d_in[10];
  const float* at_b1 = (const float*)d_in[11];
  const float* at_g  = (const float*)d_in[12];
  const float* at_be = (const float*)d_in[13];
  const float* at_w2 = (const float*)d_in[14];
  const float* at_b2 = (const float*)d_in[15];

  float* ws     = (float*)d_ws;
  float* stats1 = ws;          // 28
  float* g1s    = ws + 28;     // 128
  float* g1t    = ws + 156;    // 128
  float* stats2 = ws + 284;    // 256
  float* g2s    = ws + 540;    // 128
  float* g2t    = ws + 668;    // 128
  float* outp   = (float*)d_out;

  hipMemsetAsync(d_ws, 0, 796 * sizeof(float), stream);

  k_moments<<<(Bq*Mq*Kq + 255)/256, 256, 0, stream>>>(q_xyzs, k_xyzs, knn, stats1);
  k_gn1<<<1, 128, 0, stream>>>(stats1, pe_w1, pe_b1, pe_g, pe_be, g1s, g1t);

  size_t shBase = (size_t)3*64*72*2 + (256 + 64*8 + 128)*sizeof(float);
  size_t sh2 = shBase + 8*(size_t)(2*16*72*2);
  size_t shF = shBase + 8*(size_t)(2*16*72*2 + 2*64*18*4);

  k_pipe<false><<<2*GRIDB, 256, sh2, stream>>>(q_xyzs, k_xyzs, knn, mask,
      pe_w1, pe_b1, pe_w2, pe_b2, at_w1, at_b1, at_w2, at_b2,
      g1s, g1t, g2s, g2t, stats2);
  k_gn2<<<1, 128, 0, stream>>>(stats2, at_g, at_be, g2s, g2t);
  k_pipe<true><<<2*GRIDB, 256, shF, stream>>>(q_xyzs, k_xyzs, knn, mask,
      pe_w1, pe_b1, pe_w2, pe_b2, at_w1, at_b1, at_w2, at_b2,
      g1s, g1t, g2s, g2t, outp);

  (void)in_sizes; (void)n_in; (void)out_size; (void)ws_size;
}
